// Inference_19335942766763
// MI455X (gfx1250) — matrix operand order repaired and checked by reading, not compiled, not run on hardware
//
#include <hip/hip_runtime.h>
#include <hip/hip_bf16.h>
#include <stdint.h>

// Problem constants (from the reference)
#define BB 32     // batch
#define TT 2048   // time steps
#define DD 640    // DE = DP = DJ
#define VV 1024   // vocab
#define G4 2560   // 4*DP (LSTM gates)
#define KT32 20   // 640 / 32 (K-steps per 16x16x32 WMMA)

typedef __attribute__((ext_vector_type(16))) __bf16 v16bf;
typedef __attribute__((ext_vector_type(8)))  __bf16 v8bf;
typedef __attribute__((ext_vector_type(8)))  float  v8f;
typedef __attribute__((ext_vector_type(4)))  float  v4f;

// D = A(16x32 bf16) * B(32x16 bf16) + C(16x16 f32), wave32
__device__ inline v8f wmma_bf16(v16bf a, v16bf b, v8f c) {
  return __builtin_amdgcn_wmma_f32_16x16x32_bf16(false, a, false, b, (short)0, c,
                                                 false, false);
}

// A-fragment: lane holds row M=lane%16; K(i) = (i<8?0:16) + (lane>>4)*8 + (i&7).
// p points at row*lda + (lane>>4)*8 (+ kt*32); two contiguous 16B loads.
__device__ inline v16bf load_a_bf(const __bf16* p) {
  v8bf lo = *(const v8bf*)(p);
  v8bf hi = *(const v8bf*)(p + 16);
  return __builtin_shufflevector(lo, hi, 0, 1, 2, 3, 4, 5, 6, 7, 8, 9, 10, 11,
                                 12, 13, 14, 15);
}

// Same A-fragment, but converting from f32 rows on the fly (used once for P).
__device__ inline v16bf load_a_f32(const float* p) {
  v4f a0 = *(const v4f*)(p);
  v4f a1 = *(const v4f*)(p + 4);
  v4f a2 = *(const v4f*)(p + 16);
  v4f a3 = *(const v4f*)(p + 20);
  v16bf r;
#pragma unroll
  for (int i = 0; i < 4; ++i) {
    r[i]      = (__bf16)a0[i];
    r[i + 4]  = (__bf16)a1[i];
    r[i + 8]  = (__bf16)a2[i];
    r[i + 12] = (__bf16)a3[i];
  }
  return r;
}

__device__ inline float sigm(float x) { return 1.f / (1.f + __expf(-x)); }

// Generation-counter grid barrier (all blocks call it the same number of times).
__device__ inline void gsync(unsigned* bar, unsigned nblk) {
  __syncthreads();
  if (threadIdx.x == 0) {
    __threadfence();
    unsigned old = __hip_atomic_fetch_add(bar, 1u, __ATOMIC_ACQ_REL,
                                          __HIP_MEMORY_SCOPE_AGENT);
    unsigned tgt = (old / nblk + 1u) * nblk;
    while (__hip_atomic_load(bar, __ATOMIC_ACQUIRE, __HIP_MEMORY_SCOPE_AGENT) < tgt)
      __builtin_amdgcn_s_sleep(2);
  }
  __syncthreads();
}

// ---- init: h/c from states[1,2,B,Dp], last=BLANK, bf16 stage of h, barrier=0
__global__ void k_init(const float* __restrict__ states, float* h, float* c,
                       unsigned short* h_bfu, int* last, unsigned* bar) {
  __bf16* h_bf = (__bf16*)h_bfu;
  int i = blockIdx.x * blockDim.x + threadIdx.x;
  if (i < BB * DD) {
    float hv = states[i];
    float cv = states[BB * DD + i];
    h[i] = hv;
    c[i] = cv;
    h_bf[i] = (__bf16)hv;
  }
  if (i < BB) last[i] = 0;
  if (i < 16) bar[i] = 0u;
}

// ---- pack f32 weight W[K][N] into WMMA bf16 B-fragments.
// Fragment f = nt*(K/32)+kt; per lane 16 bf16: K = kt*32 + (i<8?0:16) + (lane>>4)*8 + (i&7),
// N = nt*16 + (lane&15). One contiguous 32B store per lane.
__global__ void k_pack(const float* __restrict__ src, unsigned short* dstu,
                       int K, int N) {
  __bf16* dst = (__bf16*)dstu;
  int gid = blockIdx.x * blockDim.x + threadIdx.x;
  int KT = K >> 5, NT = N >> 4;
  if (gid >= KT * NT * 32) return;
  int lane = gid & 31, f = gid >> 5;
  int nt = f / KT, kt = f - nt * KT;
  int n = nt * 16 + (lane & 15);
  int kb = kt * 32 + ((lane >> 4) << 3);
  v16bf r;
#pragma unroll
  for (int i = 0; i < 16; ++i) r[i] = (__bf16)src[(size_t)(kb + i + (i & 8)) * N + n];
  *(v16bf*)(dst + (size_t)gid * 16) = r;
}

// ---- P = embed @ Wx + b_lstm   (V x 4Dp), fully parallel WMMA GEMM
__global__ void __launch_bounds__(256) k_pgemm(const float* __restrict__ embed,
                                               const unsigned short* wxpku,
                                               const float* __restrict__ b_lstm,
                                               float* __restrict__ P) {
  const __bf16* wxpk = (const __bf16*)wxpku;
  const int lane = threadIdx.x & 31;
  const int wid = (blockIdx.x * blockDim.x + threadIdx.x) >> 5;
  const int nwaves = (gridDim.x * blockDim.x) >> 5;
  const int MT = VV / 16, NT = G4 / 16;  // 64 x 160 tiles
  for (int tile = wid; tile < MT * NT; tile += nwaves) {
    int mt = tile / NT, nt = tile - mt * NT;
    v8f acc = {0.f, 0.f, 0.f, 0.f, 0.f, 0.f, 0.f, 0.f};
    const float* arow =
        embed + (size_t)(mt * 16 + (lane & 15)) * DD + ((lane >> 4) << 3);
    const __bf16* bfr = wxpk + (size_t)(nt * KT32) * 512 + lane * 16;
    for (int kt = 0; kt < KT32; ++kt)
      acc = wmma_bf16(load_a_f32(arow + kt * 32),
                      *(const v16bf*)(bfr + kt * 512), acc);
    int n = nt * 16 + (lane & 15);
    int mrow = mt * 16 + ((lane >> 4) << 3);
#pragma unroll
    for (int v = 0; v < 8; ++v)
      P[(size_t)(mrow + v) * G4 + n] = acc[v] + b_lstm[n];
  }
}

// ---- persistent decoder: 2048 serial steps, 5 phases per step, grid-synced
__global__ void __launch_bounds__(256) k_decode(
    const float* __restrict__ enc, const int* __restrict__ lens,
    const float* __restrict__ P, const unsigned short* whpku,
    const unsigned short* wepku, const unsigned short* wppku,
    const unsigned short* wvpku, const float* __restrict__ b_j,
    const float* __restrict__ b_v, float* h, float* c, float* hn, float* cn,
    unsigned short* h_bfu, unsigned short* hn_bfu, unsigned short* enc_bfu,
    unsigned short* z_bfu, float* gates, float* logits, int* last,
    unsigned* bar, int* out_labels, float* out_scores) {
  const __bf16* whpk = (const __bf16*)whpku;
  const __bf16* wepk = (const __bf16*)wepku;
  const __bf16* wppk = (const __bf16*)wppku;
  const __bf16* wvpk = (const __bf16*)wvpku;
  __bf16* h_bf = (__bf16*)h_bfu;
  __bf16* hn_bf = (__bf16*)hn_bfu;
  __bf16* enc_bf = (__bf16*)enc_bfu;
  __bf16* z_bf = (__bf16*)z_bfu;

  __shared__ float smax[256];
  __shared__ int sarg[256];
  __shared__ float ssum[256];

  const int lane = threadIdx.x & 31;
  const int gtid = blockIdx.x * blockDim.x + threadIdx.x;
  const int gthreads = gridDim.x * blockDim.x;
  const int wid = gtid >> 5;
  const int nwaves = gthreads >> 5;
  const unsigned nblk = gridDim.x;

  for (int t = 0; t < TT; ++t) {
    // Phase 1: gates = P[last] + h @ Wh   (M=32, N=2560, K=640)
    for (int tile = wid; tile < 2 * 160; tile += nwaves) {
      int mt = tile / 160, nt = tile - mt * 160;
      v8f acc = {0.f, 0.f, 0.f, 0.f, 0.f, 0.f, 0.f, 0.f};
      const __bf16* arow =
          h_bf + (mt * 16 + (lane & 15)) * DD + ((lane >> 4) << 3);
      const __bf16* bfr = whpk + (size_t)(nt * KT32) * 512 + lane * 16;
      for (int kt = 0; kt < KT32; ++kt) {
        __builtin_prefetch((const void*)(bfr + (kt + 1) * 512), 0, 1);
        acc = wmma_bf16(load_a_bf(arow + kt * 32),
                        *(const v16bf*)(bfr + kt * 512), acc);
      }
      int n = nt * 16 + (lane & 15);
      int mrow = mt * 16 + ((lane >> 4) << 3);
#pragma unroll
      for (int v = 0; v < 8; ++v) {
        int m = mrow + v;
        gates[(size_t)m * G4 + n] = acc[v] + P[(size_t)last[m] * G4 + n];
      }
    }
    gsync(bar, nblk);

    // Phase 2: LSTM pointwise; stage h_new and enc_t as bf16 A-operands
    for (int i = gtid; i < BB * DD; i += gthreads) {
      int b = i / DD, d = i - b * DD;
      const float* g = gates + (size_t)b * G4 + d;
      float ig = g[0], fg = g[DD], gg = g[2 * DD], og = g[3 * DD];
      float cv = sigm(fg) * c[i] + sigm(ig) * tanhf(gg);
      float hv = sigm(og) * tanhf(cv);
      cn[i] = cv;
      hn[i] = hv;
      hn_bf[i] = (__bf16)hv;
      enc_bf[i] = (__bf16)enc[((size_t)b * TT + t) * DD + d];
    }
    gsync(bar, nblk);

    // Phase 3: z = tanh(enc_t @ We + h_new @ Wp + b_j)   (M=32, N=640)
    for (int tile = wid; tile < 2 * 40; tile += nwaves) {
      int mt = tile / 40, nt = tile - mt * 40;
      v8f acc = {0.f, 0.f, 0.f, 0.f, 0.f, 0.f, 0.f, 0.f};
      const __bf16* arowE =
          enc_bf + (mt * 16 + (lane & 15)) * DD + ((lane >> 4) << 3);
      const __bf16* arowH =
          hn_bf + (mt * 16 + (lane & 15)) * DD + ((lane >> 4) << 3);
      const __bf16* bfe = wepk + (size_t)(nt * KT32) * 512 + lane * 16;
      const __bf16* bfp = wppk + (size_t)(nt * KT32) * 512 + lane * 16;
      for (int kt = 0; kt < KT32; ++kt)
        acc = wmma_bf16(load_a_bf(arowE + kt * 32),
                        *(const v16bf*)(bfe + kt * 512), acc);
      for (int kt = 0; kt < KT32; ++kt)
        acc = wmma_bf16(load_a_bf(arowH + kt * 32),
                        *(const v16bf*)(bfp + kt * 512), acc);
      int n = nt * 16 + (lane & 15);
      int mrow = mt * 16 + ((lane >> 4) << 3);
#pragma unroll
      for (int v = 0; v < 8; ++v)
        z_bf[(mrow + v) * DD + n] = (__bf16)tanhf(acc[v] + b_j[n]);
    }
    gsync(bar, nblk);

    // Phase 4: logits = z @ Wv + b_v   (M=32, N=1024)
    for (int tile = wid; tile < 2 * 64; tile += nwaves) {
      int mt = tile / 64, nt = tile - mt * 64;
      v8f acc = {0.f, 0.f, 0.f, 0.f, 0.f, 0.f, 0.f, 0.f};
      const __bf16* arow =
          z_bf + (mt * 16 + (lane & 15)) * DD + ((lane >> 4) << 3);
      const __bf16* bfr = wvpk + (size_t)(nt * KT32) * 512 + lane * 16;
      for (int kt = 0; kt < KT32; ++kt) {
        __builtin_prefetch((const void*)(bfr + (kt + 1) * 512), 0, 1);
        acc = wmma_bf16(load_a_bf(arow + kt * 32),
                        *(const v16bf*)(bfr + kt * 512), acc);
      }
      int n = nt * 16 + (lane & 15);
      int mrow = mt * 16 + ((lane >> 4) << 3);
#pragma unroll
      for (int v = 0; v < 8; ++v)
        logits[(mrow + v) * VV + n] = acc[v] + b_v[n];
    }
    gsync(bar, nblk);

    // Phase 5: per-row argmax + log-sum-exp + masked state/label update
    if (blockIdx.x < BB) {
      int b = blockIdx.x;
      float bmax = -3.4e38f;
      int barg = 0;
      for (int vv = threadIdx.x; vv < VV; vv += 256) {
        float x = logits[b * VV + vv];
        if (x > bmax) { bmax = x; barg = vv; }
      }
      smax[threadIdx.x] = bmax;
      sarg[threadIdx.x] = barg;
      __syncthreads();
      for (int s = 128; s > 0; s >>= 1) {
        if (threadIdx.x < s) {
          float xo = smax[threadIdx.x + s];
          int ao = sarg[threadIdx.x + s];
          if (xo > smax[threadIdx.x] ||
              (xo == smax[threadIdx.x] && ao < sarg[threadIdx.x])) {
            smax[threadIdx.x] = xo;
            sarg[threadIdx.x] = ao;
          }
        }
        __syncthreads();
      }
      float mx = smax[0];
      int sym = sarg[0];
      float lsum = 0.f;
      for (int vv = threadIdx.x; vv < VV; vv += 256)
        lsum += __expf(logits[b * VV + vv] - mx);
      ssum[threadIdx.x] = lsum;
      __syncthreads();
      for (int s = 128; s > 0; s >>= 1) {
        if (threadIdx.x < s) ssum[threadIdx.x] += ssum[threadIdx.x + s];
        __syncthreads();
      }
      bool msk = (t >= lens[b]) || (sym == 0);
      if (threadIdx.x == 0) {
        out_labels[(size_t)b * TT + t] = msk ? 0 : sym;
        out_scores[(size_t)b * TT + t] = -__logf(ssum[0]);  // ytu[b, argmax]
        if (!msk) last[b] = sym;
      }
      for (int d = threadIdx.x; d < DD; d += 256) {
        int i = b * DD + d;
        float hv = msk ? h[i] : hn[i];
        float cv = msk ? c[i] : cn[i];
        h[i] = hv;
        c[i] = cv;
        h_bf[i] = (__bf16)hv;
      }
    }
    gsync(bar, nblk);
  }
}

extern "C" void kernel_launch(void* const* d_in, const int* in_sizes, int n_in,
                              void* d_out, int out_size, void* d_ws,
                              size_t ws_size, hipStream_t stream) {
  const float* enc = (const float*)d_in[0];
  const int* lens = (const int*)d_in[1];
  const float* states = (const float*)d_in[2];
  const float* embed = (const float*)d_in[3];
  const float* Wx = (const float*)d_in[4];
  const float* Wh = (const float*)d_in[5];
  const float* b_lstm = (const float*)d_in[6];
  const float* We = (const float*)d_in[7];
  const float* Wp = (const float*)d_in[8];
  const float* b_j = (const float*)d_in[9];
  const float* Wv = (const float*)d_in[10];
  const float* b_v = (const float*)d_in[11];
  (void)in_sizes; (void)n_in; (void)out_size; (void)ws_size;

  char* w = (char*)d_ws;
  size_t off = 0;
  auto take = [&](size_t bytes) -> void* {
    void* p = w + off;
    off = (off + bytes + 4095) & ~(size_t)4095;
    return p;
  };
  float* P = (float*)take((size_t)VV * G4 * 4);                         // 10.5 MB
  unsigned short* wxpk = (unsigned short*)take((size_t)(G4 / 16) * KT32 * 512 * 2);
  unsigned short* whpk = (unsigned short*)take((size_t)(G4 / 16) * KT32 * 512 * 2);
  unsigned short* wepk = (unsigned short*)take((size_t)(DD / 16) * KT32 * 512 * 2);
  unsigned short* wppk = (unsigned short*)take((size_t)(DD / 16) * KT32 * 512 * 2);
  unsigned short* wvpk = (unsigned short*)take((size_t)(VV / 16) * KT32 * 512 * 2);
  float* h = (float*)take((size_t)BB * DD * 4);
  float* c = (float*)take((size_t)BB * DD * 4);
  float* hn = (float*)take((size_t)BB * DD * 4);
  float* cn = (float*)take((size_t)BB * DD * 4);
  unsigned short* h_bf = (unsigned short*)take((size_t)BB * DD * 2);
  unsigned short* hn_bf = (unsigned short*)take((size_t)BB * DD * 2);
  unsigned short* enc_bf = (unsigned short*)take((size_t)BB * DD * 2);
  unsigned short* z_bf = (unsigned short*)take((size_t)BB * DD * 2);
  float* gates = (float*)take((size_t)BB * G4 * 4);
  float* logits = (float*)take((size_t)BB * VV * 4);
  int* last = (int*)take(BB * 4);
  unsigned* bar = (unsigned*)take(256);

  k_init<<<(BB * DD + 255) / 256, 256, 0, stream>>>(states, h, c, h_bf, last, bar);

  // one-time weight repack into WMMA bf16 B-fragment layout (lives in L2)
  k_pack<<<((G4 / 16) * KT32 * 32 + 255) / 256, 256, 0, stream>>>(Wx, wxpk, DD, G4);
  k_pack<<<((G4 / 16) * KT32 * 32 + 255) / 256, 256, 0, stream>>>(Wh, whpk, DD, G4);
  k_pack<<<((DD / 16) * KT32 * 32 + 255) / 256, 256, 0, stream>>>(We, wepk, DD, DD);
  k_pack<<<((DD / 16) * KT32 * 32 + 255) / 256, 256, 0, stream>>>(Wp, wppk, DD, DD);
  k_pack<<<((VV / 16) * KT32 * 32 + 255) / 256, 256, 0, stream>>>(Wv, wvpk, DD, VV);

  // hoist embedding projection off the serial path: P = embed @ Wx + b_lstm
  k_pgemm<<<512, 256, 0, stream>>>(embed, wxpk, b_lstm, P);

  int* out_labels = (int*)d_out;                    // labels.T  [B,T] int32
  float* out_scores = (float*)d_out + (size_t)BB * TT;  // scores.T [B,T] f32

  k_decode<<<64, 256, 0, stream>>>(enc, lens, P, whpk, wepk, wppk, wvpk, b_j,
                                   b_v, h, c, hn, cn, h_bf, hn_bf, enc_bf,
                                   z_bf, gates, logits, last, bar, out_labels,
                                   out_scores);
}
